// DifferentiableNeuralComputer_46162308497798
// MI455X (gfx1250) — compile-verified, matrix operand order checked by reading, not run on hardware
//
#include <hip/hip_runtime.h>
#include <hip/hip_bf16.h>
#include <math.h>

// Problem dims (match reference)
#define B_   16
#define T_   512
#define I_   256
#define H_   512
#define M_   2048
#define D_   64
#define NH_  4
#define O_   256
#define G4H_ (4 * H_)         // 2048
#define CW_  (H_ + NH_ * D_)  // 768 combined width
#define EPSF 1e-8f

typedef __bf16 bf16_t;
typedef __attribute__((ext_vector_type(8)))  bf16_t v8bf;
typedef __attribute__((ext_vector_type(16))) bf16_t v16bf;
typedef __attribute__((ext_vector_type(8)))  float  v8f;

// ---------------------------------------------------------------------------
// WMMA helpers (CDNA5 16x16x32 bf16, f32 accumulate)
// A-matrix 16x32 layout: lane L holds row M = L&15; lanes 0-15 carry
// K = {k0..k0+7, k0+16..k0+23}, lanes 16-31 carry K = {k0+8.., k0+24..}.
// B-matrix 32x16 is the mirrored layout (lane = N column), so a row-major
// weight W[N,K] tile loads with the identical addressing.
// ---------------------------------------------------------------------------
__device__ __forceinline__ v16bf load_tile16x32(const bf16_t* p,
                                                int ld, int row, int k0, int half) {
  const bf16_t* base = p + (size_t)row * ld + k0 + half * 8;
  v8bf lo = *(const v8bf*)(base);        // K = k0 + half*8 .. +7
  v8bf hi = *(const v8bf*)(base + 16);   // K = k0 + 16 + half*8 .. +7
  return __builtin_shufflevector(lo, hi, 0, 1, 2, 3, 4, 5, 6, 7,
                                         8, 9, 10, 11, 12, 13, 14, 15);
}

__device__ __forceinline__ v8f wmma_bf16(v16bf a, v16bf b, v8f c) {
  return __builtin_amdgcn_wmma_f32_16x16x32_bf16(
      /*neg_a=*/false, a, /*neg_b=*/false, b,
      /*c_mod=*/(short)0, c, /*reuse_a=*/false, /*reuse_b=*/false);
}

__device__ __forceinline__ float sigm(float x) { return 1.0f / (1.0f + __expf(-x)); }

// Async global->LDS copy of one 16-byte packet (CDNA5 VGLOBAL opcode 98).
// dst_lds = LDS byte offset (addr[31:0] of the generic shared pointer),
// src     = per-lane global address. Tracked by ASYNCcnt.
__device__ __forceinline__ void async_copy_b128(unsigned dst_lds, const bf16_t* src) {
  asm volatile("global_load_async_to_lds_b128 %0, %1, off"
               :: "v"(dst_lds), "v"(src) : "memory");
}
__device__ __forceinline__ void wait_asynccnt0() {
  asm volatile("s_wait_asynccnt 0x0" ::: "memory");
}

// ---------------------------------------------------------------------------
// fp32 -> bf16 conversion (grid-stride)
// ---------------------------------------------------------------------------
__global__ void k_cvt(const float* __restrict__ in, bf16_t* __restrict__ out, int n) {
  for (int i = blockIdx.x * blockDim.x + threadIdx.x; i < n;
       i += gridDim.x * blockDim.x)
    out[i] = (bf16_t)in[i];
}

// Zero-init recurrent state
__global__ void k_init_state(bf16_t* hA, bf16_t* hB, float* c) {
  int i = blockIdx.x * blockDim.x + threadIdx.x;
  if (i < B_ * H_) { hA[i] = (bf16_t)0.0f; hB[i] = (bf16_t)0.0f; c[i] = 0.0f; }
}

// ---------------------------------------------------------------------------
// memory prep: bf16 copy, bf16 transpose [D,M], row norms. One wave per row.
// ---------------------------------------------------------------------------
__global__ void k_mem_prep(const float* __restrict__ mem, bf16_t* __restrict__ memb,
                           bf16_t* __restrict__ memT, float* __restrict__ mn) {
  int wid  = (blockIdx.x * blockDim.x + threadIdx.x) >> 5;
  int lane = threadIdx.x & 31;
  if (wid >= M_) return;
  float v0 = mem[(size_t)wid * D_ + lane];
  float v1 = mem[(size_t)wid * D_ + 32 + lane];
  memb[(size_t)wid * D_ + lane]        = (bf16_t)v0;
  memb[(size_t)wid * D_ + 32 + lane]   = (bf16_t)v1;
  memT[(size_t)lane * M_ + wid]        = (bf16_t)v0;
  memT[(size_t)(lane + 32) * M_ + wid] = (bf16_t)v1;
  float s = v0 * v0 + v1 * v1;
#pragma unroll
  for (int m = 1; m < 32; m <<= 1) s += __shfl_xor(s, m);
  if (lane == 0) mn[wid] = sqrtf(s);
}

// key norms + bf16 keys. One thread per (b,t,nh) row of 64.
__global__ void k_keynorm(const float* __restrict__ keys, bf16_t* __restrict__ keysb,
                          float* __restrict__ kn, int nrows) {
  int r = blockIdx.x * blockDim.x + threadIdx.x;
  if (r >= nrows) return;
  float s = 0.0f;
#pragma unroll 8
  for (int d = 0; d < D_; ++d) {
    float v = keys[(size_t)r * D_ + d];
    s += v * v;
    keysb[(size_t)r * D_ + d] = (bf16_t)v;
  }
  kn[r] = sqrtf(s);
}

// ---------------------------------------------------------------------------
// Generic bf16 GEMM: C[rows,ncols] = A[rows,K] * W[ncols,K]^T + bias0(+bias1)
// Block = 8 waves sharing an A panel (mtpb x 16 rows x K) that is staged into
// LDS once via GLOBAL_LOAD_ASYNC_TO_LDS_B128 (ASYNCcnt), then each wave runs a
// software-pipelined k-loop: A tiles from LDS (ds_load_b128), B stream from
// global, 4 accumulators per wave (16x64 strip).
// ---------------------------------------------------------------------------
__global__ void k_gemm_bf16(const bf16_t* __restrict__ A, int lda,
                            const bf16_t* __restrict__ W, int ldw,
                            const float* __restrict__ bias0,
                            const float* __restrict__ bias1,
                            float* __restrict__ C, int ldc,
                            int rows, int ncols, int K) {
  extern __shared__ __align__(16) bf16_t smem[];
  (void)rows;
  const int tid  = threadIdx.x;
  const int w    = tid >> 5;
  const int lane = tid & 31;
  const int half = lane >> 4;
  const int l16  = lane & 15;

  const int ngrp  = ncols >> 6;              // 64-wide column groups
  const int wpm   = (ngrp < 8) ? ngrp : 8;   // waves along N per block
  const int mtpb  = 8 / wpm;                 // 16-row m-tiles per block
  const int bnCnt = ngrp / wpm;
  const int bm = blockIdx.x / bnCnt;
  const int bn = blockIdx.x % bnCnt;
  const int mt = w / wpm;
  const int ng = bn * wpm + (w % wpm);
  const int mi = bm * mtpb + mt;

  // ---- cooperative async stage of the A panel into LDS ----
  const unsigned lds_base = (unsigned)(unsigned long long)(const void*)smem;
  const int nvec  = (mtpb * 16 * K) >> 3;    // # of 16-byte packets
  const int rowg0 = bm * mtpb * 16;
  for (int v = tid; v < nvec; v += 256) {
    int e   = v << 3;
    int row = e / K;
    int col = e - row * K;
    async_copy_b128(lds_base + (unsigned)(e * 2),
                    A + (size_t)(rowg0 + row) * lda + col);
  }
  wait_asynccnt0();
  __syncthreads();

  const bf16_t* Al = smem;            // A panel in LDS, row-major, ld = K
  const int arl = mt * 16 + l16;

  v8f acc[4] = {v8f{}, v8f{}, v8f{}, v8f{}};

  // prologue: tiles for k0 = 0
  v16bf a = load_tile16x32(Al, K, arl, 0, half);
  v16bf b[4];
#pragma unroll
  for (int t = 0; t < 4; ++t)
    b[t] = load_tile16x32(W, ldw, ng * 64 + t * 16 + l16, 0, half);

  int k0 = 0;
  while (k0 + 32 < K) {
    const int kn_ = k0 + 32;
    // issue next-step loads first so they overlap current WMMAs
    v16bf an = load_tile16x32(Al, K, arl, kn_, half);
    v16bf bn_[4];
#pragma unroll
    for (int t = 0; t < 4; ++t)
      bn_[t] = load_tile16x32(W, ldw, ng * 64 + t * 16 + l16, kn_, half);
#pragma unroll
    for (int t = 0; t < 4; ++t) acc[t] = wmma_bf16(a, b[t], acc[t]);
    a = an;
#pragma unroll
    for (int t = 0; t < 4; ++t) b[t] = bn_[t];
    k0 = kn_;
  }
#pragma unroll
  for (int t = 0; t < 4; ++t) acc[t] = wmma_bf16(a, b[t], acc[t]);

#pragma unroll
  for (int t = 0; t < 4; ++t) {
    int col = ng * 64 + t * 16 + l16;
    float bb = bias0[col];
    if (bias1) bb += bias1[col];
#pragma unroll
    for (int r = 0; r < 8; ++r) {
      int row = mi * 16 + r + 8 * half;
      C[(size_t)row * ldc + col] = acc[t][r] + bb;
    }
  }
}

// ---------------------------------------------------------------------------
// One LSTM step: gates[16,2048] = h_prev[16,512] @ Whh^T (+ precomputed x-gates
// incl. both biases), activations, c/h update. 32 waves, one per 16-wide j-tile,
// each wave does 4 gate tiles (i,f,g,o) for the same hidden columns.
// Software-pipelined K loop (critical latency path); x-gate lines prefetched.
// ---------------------------------------------------------------------------
__global__ void k_lstm_step(const bf16_t* __restrict__ hprev,
                            const bf16_t* __restrict__ Whh,
                            const float* __restrict__ xg,
                            float* __restrict__ cst,
                            bf16_t* __restrict__ hnext,
                            bf16_t* __restrict__ combined, int t) {
  const int wid  = (blockIdx.x * blockDim.x + threadIdx.x) >> 5;  // 0..31
  const int lane = threadIdx.x & 31;
  const int half = lane >> 4;
  const int l16  = lane & 15;
  const int j0   = wid * 16;
  const int j    = j0 + l16;

  // warm the x-gate cachelines consumed in the epilogue (overlaps the GEMM)
#pragma unroll
  for (int r = 0; r < 8; ++r) {
    size_t xrow = ((size_t)(r + 8 * half) * T_ + t) * G4H_;
    __builtin_prefetch(xg + xrow + j, 0, 3);
  }

  v8f acc[4] = {v8f{}, v8f{}, v8f{}, v8f{}};

  v16bf a = load_tile16x32(hprev, H_, l16, 0, half);  // rows = batch 0..15
  v16bf b[4];
#pragma unroll
  for (int g = 0; g < 4; ++g)
    b[g] = load_tile16x32(Whh, H_, g * H_ + j0 + l16, 0, half);

  int k0 = 0;
  while (k0 + 32 < H_) {
    const int kn_ = k0 + 32;
    v16bf an = load_tile16x32(hprev, H_, l16, kn_, half);
    v16bf bn_[4];
#pragma unroll
    for (int g = 0; g < 4; ++g)
      bn_[g] = load_tile16x32(Whh, H_, g * H_ + j0 + l16, kn_, half);
#pragma unroll
    for (int g = 0; g < 4; ++g) acc[g] = wmma_bf16(a, b[g], acc[g]);
    a = an;
#pragma unroll
    for (int g = 0; g < 4; ++g) b[g] = bn_[g];
    k0 = kn_;
  }
#pragma unroll
  for (int g = 0; g < 4; ++g) acc[g] = wmma_bf16(a, b[g], acc[g]);

#pragma unroll
  for (int r = 0; r < 8; ++r) {
    int bi = r + 8 * half;
    size_t xrow = ((size_t)bi * T_ + t) * G4H_;
    float gi = sigm(acc[0][r] + xg[xrow + j]);
    float gf = sigm(acc[1][r] + xg[xrow + H_ + j]);
    float gg = tanhf(acc[2][r] + xg[xrow + 2 * H_ + j]);
    float go = sigm(acc[3][r] + xg[xrow + 3 * H_ + j]);
    float cn = gf * cst[bi * H_ + j] + gi * gg;
    cst[bi * H_ + j] = cn;
    float hv = go * tanhf(cn);
    hnext[bi * H_ + j] = (bf16_t)hv;
    combined[((size_t)bi * T_ + t) * CW_ + j] = (bf16_t)hv;  // hs (bf16) in place
  }
}

// ---------------------------------------------------------------------------
// Flash-style cosine attention over memory. One wave owns 16 key rows,
// streams M_=2048 memory rows in chunks of 32 with online softmax.
// Next chunk's S-operand tiles are prefetched into regs during softmax math.
// ---------------------------------------------------------------------------
__global__ void k_attn(const bf16_t* __restrict__ keysb,
                       const float* __restrict__ kn,
                       const bf16_t* __restrict__ memb,   // [M, D]
                       const bf16_t* __restrict__ memT,   // [D, M]
                       const float* __restrict__ mn,
                       bf16_t* __restrict__ combined) {
  __shared__ __align__(16) bf16_t pb[4][16 * 32];  // per-wave P-tile transpose
  const int wlb  = threadIdx.x >> 5;
  const int wid  = (blockIdx.x * blockDim.x + threadIdx.x) >> 5;  // 0..2047
  const int lane = threadIdx.x & 31;
  const int half = lane >> 4;
  const int l16  = lane & 15;
  const int rb   = wid * 16;  // first key-row of this wave

  v16bf a0 = load_tile16x32(keysb, D_, rb + l16, 0, half);
  v16bf a1 = load_tile16x32(keysb, D_, rb + l16, 32, half);

  float kn8[8], m8[8], l8[8];
#pragma unroll
  for (int r = 0; r < 8; ++r) {
    kn8[r] = kn[rb + r + 8 * half];
    m8[r]  = -__builtin_inff();
    l8[r]  = 0.0f;
  }
  v8f o[4] = {v8f{}, v8f{}, v8f{}, v8f{}};

  // prologue loads for j0 = 0
  v16bf b00 = load_tile16x32(memb, D_, 0 + l16, 0, half);
  v16bf b01 = load_tile16x32(memb, D_, 0 + l16, 32, half);
  v16bf b10 = load_tile16x32(memb, D_, 16 + l16, 0, half);
  v16bf b11 = load_tile16x32(memb, D_, 16 + l16, 32, half);

  for (int j0 = 0; j0 < M_; j0 += 32) {
    // S = K @ mem_chunk^T   (16 x 32 scores)
    v8f s0 = {}, s1 = {};
    s0 = wmma_bf16(a0, b00, s0);
    s0 = wmma_bf16(a1, b01, s0);
    s1 = wmma_bf16(a0, b10, s1);
    s1 = wmma_bf16(a1, b11, s1);

    // prefetch next chunk's S operands; loads fly during the softmax math
    const bool more = (j0 + 32 < M_);
    v16bf n00 = b00, n01 = b01, n10 = b10, n11 = b11;
    if (more) {
      n00 = load_tile16x32(memb, D_, j0 + 32 + l16, 0, half);
      n01 = load_tile16x32(memb, D_, j0 + 32 + l16, 32, half);
      n10 = load_tile16x32(memb, D_, j0 + 48 + l16, 0, half);
      n11 = load_tile16x32(memb, D_, j0 + 48 + l16, 32, half);
    }

    float mn0 = mn[j0 + l16];
    float mn1 = mn[j0 + 16 + l16];
    float alpha8[8];
#pragma unroll
    for (int r = 0; r < 8; ++r) {
      float x0 = s0[r] / fmaxf(kn8[r] * mn0, EPSF);
      float x1 = s1[r] / fmaxf(kn8[r] * mn1, EPSF);
      // row max across the 16 lanes of this half-wave (wave32)
      float tm = fmaxf(x0, x1);
      tm = fmaxf(tm, __shfl_xor(tm, 1));
      tm = fmaxf(tm, __shfl_xor(tm, 2));
      tm = fmaxf(tm, __shfl_xor(tm, 4));
      tm = fmaxf(tm, __shfl_xor(tm, 8));
      float mnew = fmaxf(m8[r], tm);
      float al = __expf(m8[r] - mnew);
      float p0 = __expf(x0 - mnew);
      float p1 = __expf(x1 - mnew);
      float rs = p0 + p1;
      rs += __shfl_xor(rs, 1);
      rs += __shfl_xor(rs, 2);
      rs += __shfl_xor(rs, 4);
      rs += __shfl_xor(rs, 8);
      l8[r] = l8[r] * al + rs;
      m8[r] = mnew;
      alpha8[r] = al;
      // stash P (C-layout) into LDS for re-read in A-layout
      pb[wlb][(r + 8 * half) * 32 + l16]      = (bf16_t)p0;
      pb[wlb][(r + 8 * half) * 32 + 16 + l16] = (bf16_t)p1;
    }
#pragma unroll
    for (int dt = 0; dt < 4; ++dt)
#pragma unroll
      for (int r = 0; r < 8; ++r) o[dt][r] *= alpha8[r];

    __syncthreads();  // uniform trip count; orders LDS store -> load
    v16bf pa = load_tile16x32(&pb[wlb][0], 32, l16, 0, half);
    __syncthreads();

    // O += P @ mem_chunk  (B operand from transposed memory, contiguous b128)
#pragma unroll
    for (int dt = 0; dt < 4; ++dt) {
      v16bf bm = load_tile16x32(memT, M_, dt * 16 + l16, j0, half);
      o[dt] = wmma_bf16(pa, bm, o[dt]);
    }

    if (more) { b00 = n00; b01 = n01; b10 = n10; b11 = n11; }
  }

#pragma unroll
  for (int dt = 0; dt < 4; ++dt)
#pragma unroll
    for (int r = 0; r < 8; ++r) {
      int kr = rb + r + 8 * half;     // global key row = (bt)*NH + nh
      int bt = kr >> 2;
      int nh = kr & 3;
      int d  = dt * 16 + l16;
      float val = o[dt][r] / l8[r];
      combined[(size_t)bt * CW_ + H_ + nh * D_ + d] = (bf16_t)val;
    }
}

// ---------------------------------------------------------------------------
// Host orchestration
// ---------------------------------------------------------------------------
extern "C" void kernel_launch(void* const* d_in, const int* in_sizes, int n_in,
                              void* d_out, int out_size, void* d_ws, size_t ws_size,
                              hipStream_t stream) {
  const float* x    = (const float*)d_in[0];
  const float* mem  = (const float*)d_in[1];
  const float* Wih  = (const float*)d_in[2];
  const float* Whh  = (const float*)d_in[3];
  const float* bih  = (const float*)d_in[4];
  const float* bhh  = (const float*)d_in[5];
  const float* Wif  = (const float*)d_in[6];
  const float* bif  = (const float*)d_in[7];
  const float* Wout = (const float*)d_in[8];
  const float* bout = (const float*)d_in[9];
  float* out = (float*)d_out;
  (void)in_sizes; (void)n_in; (void)out_size; (void)ws_size;

  char* ws = (char*)d_ws;
  size_t off = 0;
  auto carve = [&](size_t bytes) -> void* {
    off = (off + 255) & ~(size_t)255;
    void* p = ws + off;
    off += bytes;
    return p;
  };
  const int BT = B_ * T_;                                    // 8192
  float*  xg    = (float*) carve((size_t)BT * G4H_ * 4);     // 64 MB x-gates
  bf16_t* comb  = (bf16_t*)carve((size_t)BT * CW_ * 2);      // [8192,768] hs|reads
  bf16_t* xb    = (bf16_t*)carve((size_t)BT * I_ * 2);
  bf16_t* Wihb  = (bf16_t*)carve((size_t)G4H_ * I_ * 2);
  bf16_t* Whhb  = (bf16_t*)carve((size_t)G4H_ * H_ * 2);
  bf16_t* Wifb  = (bf16_t*)carve((size_t)(NH_ * D_) * H_ * 2);  // first 256 rows of W_if
  bf16_t* Woutb = (bf16_t*)carve((size_t)O_ * CW_ * 2);
  bf16_t* memb  = (bf16_t*)carve((size_t)M_ * D_ * 2);
  bf16_t* memT  = (bf16_t*)carve((size_t)D_ * M_ * 2);
  float*  mn    = (float*) carve((size_t)M_ * 4);
  float*  keys  = (float*) carve((size_t)BT * (NH_ * D_) * 4);
  bf16_t* keysb = (bf16_t*)carve((size_t)BT * NH_ * D_ * 2);
  float*  kn    = (float*) carve((size_t)BT * NH_ * 4);
  bf16_t* hA    = (bf16_t*)carve((size_t)B_ * H_ * 2);
  bf16_t* hB    = (bf16_t*)carve((size_t)B_ * H_ * 2);
  float*  cst   = (float*) carve((size_t)B_ * H_ * 4);

  // 0) init + precision conversion
  k_init_state<<<32, 256, 0, stream>>>(hA, hB, cst);
  k_cvt<<<1024, 256, 0, stream>>>(x,    xb,    BT * I_);
  k_cvt<<<512,  256, 0, stream>>>(Wih,  Wihb,  G4H_ * I_);
  k_cvt<<<1024, 256, 0, stream>>>(Whh,  Whhb,  G4H_ * H_);
  k_cvt<<<256,  256, 0, stream>>>(Wif,  Wifb,  NH_ * D_ * H_);  // rows 0..255 are contiguous
  k_cvt<<<256,  256, 0, stream>>>(Wout, Woutb, O_ * CW_);
  k_mem_prep<<<256, 256, 0, stream>>>(mem, memb, memT, mn);

  // 1) x-gates: [8192,2048] = x @ W_ih^T + b_ih + b_hh
  //    8 waves/block share one 16xK A panel (8 KB LDS, async-staged)
  k_gemm_bf16<<<2048, 256, 16 * I_ * 2, stream>>>(xb, I_, Wihb, I_, bih, bhh,
                                                  xg, G4H_, BT, G4H_, I_);

  // 2) sequential LSTM recurrence (512 captured launches, 32 waves each)
  for (int t = 0; t < T_; ++t) {
    const bf16_t* hp = (t & 1) ? hB : hA;
    bf16_t*       hn = (t & 1) ? hA : hB;
    k_lstm_step<<<4, 256, 0, stream>>>(hp, Whhb, xg, cst, hn, comb, t);
  }

  // 3) read keys only (dead-code-eliminated iface): [8192,256] = hs @ W_if[:256]^T + b_if[:256]
  //    2 m-tiles/block -> 32x512 A panel (32 KB LDS, async-staged)
  k_gemm_bf16<<<256, 256, 32 * H_ * 2, stream>>>(comb, CW_, Wifb, H_, bif, nullptr,
                                                 keys, NH_ * D_, BT, NH_ * D_, H_);

  // 4) key norms + bf16 keys
  k_keynorm<<<128, 256, 0, stream>>>(keys, keysb, kn, BT * NH_);

  // 5) cosine-softmax reads, written into comb[:, 512:768]
  k_attn<<<512, 128, 0, stream>>>(keysb, kn, memb, memT, mn, comb);

  // 6) output projection: [8192,256] = combined @ W_out^T + b_out
  //    2 m-tiles/block -> 32x768 A panel (48 KB LDS, async-staged)
  k_gemm_bf16<<<256, 256, 32 * CW_ * 2, stream>>>(comb, CW_, Woutb, CW_, bout, nullptr,
                                                  out, O_, BT, O_, CW_);
}
